// GAT_layer_multihead_21629455303170
// MI455X (gfx1250) — matrix operand order repaired and checked by reading, not compiled, not run on hardware
//
#include <hip/hip_runtime.h>
#include <hip/hip_bf16.h>

#define Bq   4
#define Nn   256
#define Hh   8
#define DNI  32
#define DEI  16
#define DNO  32
#define DEO  32
#define LEAKY 0.2f
#define NEGV (-9e15f)

typedef __attribute__((ext_vector_type(16))) _Float16 v16h;
typedef __attribute__((ext_vector_type(8)))  float    v8f;

union AF { v16h v; unsigned int u[8]; };

__device__ inline unsigned int pack2h(float a, float b) {
    union { _Float16 h[2]; unsigned int u; } x;
    x.h[0] = (_Float16)a; x.h[1] = (_Float16)b;
    return x.u;
}

// ---------------------------------------------------------------------------
// Kernel 1: per-node prep.  Wh = nf@W^T+b, od = nf@Wod^T+b, Whi/Whj = Wh@a^T
// grid = B*N blocks of 256 threads (one thread per output feature)
// ---------------------------------------------------------------------------
__global__ __launch_bounds__(256) void gat_node_prep(
    const float* __restrict__ node_fea,
    const float* __restrict__ W_w,  const float* __restrict__ W_b,
    const float* __restrict__ Wod_w,const float* __restrict__ Wod_b,
    const float* __restrict__ a1_w, const float* __restrict__ a2_w,
    float* __restrict__ Wh, float* __restrict__ od,
    float* __restrict__ Whi, float* __restrict__ Whj)
{
    __shared__ float nf[256];
    __shared__ float wh_l[256];
    const int b = blockIdx.x >> 8;
    const int n = blockIdx.x & 255;
    const int t = threadIdx.x;
    const int h = t >> 5, c = t & 31;
    // nf[b,n,k] with k = h*DNI + c  <-  node_fea[b,h,n,c]
    nf[t] = node_fea[((b*Hh + h)*Nn + n)*DNI + c];
    __syncthreads();

    float aw = W_b[t], ao = Wod_b[t];
    const float* wr = W_w   + t*256;
    const float* wo = Wod_w + t*256;
    for (int k = 0; k < 256; ++k) { float x = nf[k]; aw += x*wr[k]; ao += x*wo[k]; }
    Wh[(b*Nn + n)*256 + t] = aw;
    od[(b*Nn + n)*256 + t] = ao;
    wh_l[t] = aw;
    __syncthreads();

    if (t < 16) {
        const int hh = t & 7;
        const float* a = (t < 8 ? a1_w : a2_w) + hh*256;
        float s = 0.f;
        for (int k = 0; k < 256; ++k) s += wh_l[k]*a[k];
        if (t < 8) Whi[(b*Nn + n)*8 + hh] = s;
        else       Whj[(b*Nn + n)*8 + hh] = s;
    }
}

// ---------------------------------------------------------------------------
// Kernel 2: pack We_w [256,128] and ae_w [8,128] (zero-padded to 16 cols)
// into f16 WMMA B-fragment order: frag(ct,ks): lane l -> col = ct*16 + l%16,
// VGPR v -> K = ks*32 + (l/16)*16 + 2v (+1 in high half)
// ---------------------------------------------------------------------------
__global__ __launch_bounds__(256) void gat_pack(
    const float* __restrict__ We_w, const float* __restrict__ ae_w,
    unsigned int* __restrict__ bpackWe, unsigned int* __restrict__ bpackAe)
{
    const int t = threadIdx.x;
    for (int p = t; p < 16*4*32*8; p += 256) {
        int v = p & 7, lane = (p >> 3) & 31, ks = (p >> 8) & 3, ct = p >> 10;
        int col = (ct << 4) + (lane & 15);
        int k   = (ks << 5) + ((lane >> 4) << 4) + (v << 1);
        bpackWe[p] = pack2h(We_w[col*128 + k], We_w[col*128 + k + 1]);
    }
    for (int p = t; p < 4*32*8; p += 256) {
        int v = p & 7, lane = (p >> 3) & 31, ks = p >> 8;
        int col = lane & 15;
        int k   = (ks << 5) + ((lane >> 4) << 4) + (v << 1);
        float lo = 0.f, hi = 0.f;
        if (col < 8) { lo = ae_w[col*128 + k]; hi = ae_w[col*128 + k + 1]; }
        bpackAe[p] = pack2h(lo, hi);
    }
}

// ---------------------------------------------------------------------------
// Kernel 3: fused edge-GEMM (WMMA f16) + attention logits + softmax + node out
// one workgroup (256 thr = 8 wave32) per (b, i)
// ---------------------------------------------------------------------------
__global__ __launch_bounds__(256) void gat_edge_kernel(
    const float* __restrict__ edge_fea, const int* __restrict__ adj,
    const float* __restrict__ Wh,  const float* __restrict__ od,
    const float* __restrict__ Whi, const float* __restrict__ Whj,
    const float* __restrict__ We_b, const float* __restrict__ ae_b,
    const unsigned int* __restrict__ bpackWe,
    const unsigned int* __restrict__ bpackAe,
    float* __restrict__ node_out, float* __restrict__ edge_out)
{
    __shared__ unsigned int ef_lds[128*64];   // 128 j-rows x 128 K halves (packed)
    __shared__ float e_lds[256*8];            // e / att  [j][h]
    __shared__ float whj_lds[256*8];          // Whj[b,:,:]
    __shared__ float odi_lds[256];
    __shared__ float web_lds[256];
    __shared__ int   adj_lds[256];
    __shared__ float whi_lds[8];
    __shared__ float aeb_lds[8];

    const int b = blockIdx.x >> 8;
    const int i = blockIdx.x & 255;
    const int t = threadIdx.x;
    const int lane = t & 31, wave = t >> 5;
    const int mrow = lane & 15;      // A/C column-group helpers
    const int khalf = lane >> 4;     // A K-half select, C row-half select

    // per-(b,i) preloads
    odi_lds[t] = od[(b*Nn + i)*256 + t];
    web_lds[t] = We_b[t];
    adj_lds[t] = adj[(b*Nn + i)*Nn + t];
    for (int p = t; p < 256*8; p += 256) whj_lds[p] = Whj[b*Nn*8 + p];
    if (t < 8) { whi_lds[t] = Whi[(b*Nn + i)*8 + t]; aeb_lds[t] = ae_b[t]; }

    for (int chunk = 0; chunk < 2; ++chunk) {
        __syncthreads();
        // stage ef[b, i, jbase..jbase+127, 0..127] to LDS as f16 pairs
        const int jbase = chunk*128;
        for (int p = t; p < 128*64; p += 256) {
            int row = p >> 6;             // local j
            int k   = (p & 63) << 1;      // feature index (even)
            int h   = k >> 4, c = k & 15; // k = h*DEI + c
            const float2 f2 = *(const float2*)(edge_fea +
                ((((b*Hh + h)*Nn + i)*Nn + (jbase + row))*DEI + c));
            ef_lds[p] = pack2h(f2.x, f2.y);
        }
        __syncthreads();

        // build 4 A fragments (K = 0..127) for this wave's 16-row j tile
        const int jl = wave*16 + mrow;    // lane's M-row within the chunk
        AF a[4];
        #pragma unroll
        for (int ks = 0; ks < 4; ++ks) {
            #pragma unroll
            for (int v = 0; v < 8; ++v)
                a[ks].u[v] = ef_lds[jl*64 + ks*16 + khalf*8 + v];
        }
        const int j0 = jbase + wave*16;

        // --- edge update GEMM: 16 column tiles of 16 outputs, K=128 ---
        for (int ct = 0; ct < 16; ++ct) {
            v8f acc = {};
            #pragma unroll
            for (int ks = 0; ks < 4; ++ks) {
                AF bf;
                const unsigned int* bp = bpackWe + (((ct << 2) + ks)*32 + lane)*8;
                #pragma unroll
                for (int v = 0; v < 8; ++v) bf.u[v] = bp[v];
                acc = __builtin_amdgcn_wmma_f32_16x16x32_f16(
                        false, a[ks].v, false, bf.v, (short)0, acc, false, false);
            }
            const int c = (ct << 4) + mrow;      // output feature
            const float addc = web_lds[c] + odi_lds[c];
            const int h = c >> 5, d = c & 31;
            float* outbase = edge_out + (((long)(b*Hh + h)*Nn + i)*Nn)*DEO + d;
            #pragma unroll
            for (int r = 0; r < 8; ++r) {
                const int j = j0 + (khalf << 3) + r;
                outbase[j*DEO] = acc[r] + addc + od[(b*Nn + j)*256 + c];
            }
        }

        // --- attention edge logits: ae GEMM (8 valid cols) ---
        {
            v8f acc = {};
            #pragma unroll
            for (int ks = 0; ks < 4; ++ks) {
                AF bf;
                const unsigned int* bp = bpackAe + (ks*32 + lane)*8;
                #pragma unroll
                for (int v = 0; v < 8; ++v) bf.u[v] = bp[v];
                acc = __builtin_amdgcn_wmma_f32_16x16x32_f16(
                        false, a[ks].v, false, bf.v, (short)0, acc, false, false);
            }
            if (mrow < 8) {
                const int h = mrow;
                const float base = aeb_lds[h] + whi_lds[h];
                #pragma unroll
                for (int r = 0; r < 8; ++r) {
                    const int j = j0 + (khalf << 3) + r;
                    float e = acc[r] + base + whj_lds[j*8 + h];
                    e = (e > 0.f) ? e : LEAKY*e;
                    e = (adj_lds[j] > 0) ? e : NEGV;
                    e_lds[j*8 + h] = e;
                }
            }
        }
    }
    __syncthreads();

    // --- per-head softmax over j (wave w owns head w) ---
    {
        const int h = wave;   // 8 waves == 8 heads
        float vals[8];
        float m = -3.4e38f;
        #pragma unroll
        for (int q = 0; q < 8; ++q) {
            vals[q] = e_lds[(lane + 32*q)*8 + h];
            m = fmaxf(m, vals[q]);
        }
        for (int off = 16; off > 0; off >>= 1) m = fmaxf(m, __shfl_xor(m, off, 32));
        float s = 0.f;
        #pragma unroll
        for (int q = 0; q < 8; ++q) { vals[q] = __expf(vals[q] - m); s += vals[q]; }
        for (int off = 16; off > 0; off >>= 1) s += __shfl_xor(s, off, 32);
        const float inv = 1.f / s;
        #pragma unroll
        for (int q = 0; q < 8; ++q) e_lds[(lane + 32*q)*8 + h] = vals[q]*inv;
    }
    __syncthreads();

    // --- node output: thread t -> (head h, dim d); att @ Whh + Whh ---
    {
        const int h = t >> 5, d = t & 31;
        const float* whcol = Wh + b*Nn*256 + h*32 + d;
        float acc = 0.f;
        for (int j = 0; j < 256; ++j) acc += e_lds[j*8 + h] * whcol[j*256];
        acc += whcol[i*256];
        node_out[((b*Hh + h)*Nn + i)*DNO + d] = acc;
    }
}

// ---------------------------------------------------------------------------
extern "C" void kernel_launch(void* const* d_in, const int* in_sizes, int n_in,
                              void* d_out, int out_size, void* d_ws, size_t ws_size,
                              hipStream_t stream) {
    const float* node_fea = (const float*)d_in[0];
    const float* edge_fea = (const float*)d_in[1];
    const int*   adj      = (const int*)  d_in[2];
    const float* W_w   = (const float*)d_in[3];
    const float* W_b   = (const float*)d_in[4];
    const float* a1_w  = (const float*)d_in[5];
    const float* a2_w  = (const float*)d_in[6];
    const float* ae_w  = (const float*)d_in[7];
    const float* ae_b  = (const float*)d_in[8];
    const float* We_w  = (const float*)d_in[9];
    const float* We_b  = (const float*)d_in[10];
    const float* Wod_w = (const float*)d_in[11];
    const float* Wod_b = (const float*)d_in[12];

    float* ws  = (float*)d_ws;
    float* Wh  = ws;                      // B*N*256 = 262,144 f
    float* od  = Wh  + Bq*Nn*256;         // 262,144 f
    float* Whi = od  + Bq*Nn*256;         // 8,192 f
    float* Whj = Whi + Bq*Nn*Hh;          // 8,192 f
    unsigned int* bpackWe = (unsigned int*)(Whj + Bq*Nn*Hh);  // 16,384 dw
    unsigned int* bpackAe = bpackWe + 16*4*32*8;              // 1,024 dw

    float* node_out = (float*)d_out;                 // [B,H,N,DNO]
    float* edge_out = node_out + Bq*Hh*Nn*DNO;       // [B,H,N,N,DEO]

    gat_node_prep<<<Bq*Nn, 256, 0, stream>>>(node_fea, W_w, W_b, Wod_w, Wod_b,
                                             a1_w, a2_w, Wh, od, Whi, Whj);
    gat_pack<<<1, 256, 0, stream>>>(We_w, ae_w, bpackWe, bpackAe);
    gat_edge_kernel<<<Bq*Nn, 256, 0, stream>>>(edge_fea, adj, Wh, od, Whi, Whj,
                                               We_b, ae_b, bpackWe, bpackAe,
                                               node_out, edge_out);
}
